// RealAgnosticAttResidualInteractionBlock_84129819394065
// MI455X (gfx1250) — compile-verified, matrix operand order checked by reading, not run on hardware
//
#include <hip/hip_runtime.h>
#include <cstdint>

#define N_NODES 10000
#define N_EDGES 160000
#define NP      10016        // nodes padded to multiple of 16
#define CHN     128

typedef __attribute__((ext_vector_type(16))) __bf16 v16bf;
typedef __attribute__((ext_vector_type(8)))  float  v8f;

union FragU { uint4 u[2]; v16bf v; };

__device__ __forceinline__ unsigned short f2bf(float f) {
    unsigned u = __float_as_uint(f);
    u += 0x7fffu + ((u >> 16) & 1u);          // round-to-nearest-even
    return (unsigned short)(u >> 16);
}
__device__ __forceinline__ float siluf(float x) { return x / (1.f + __expf(-x)); }

// Load a 16x32 bf16 A/B fragment from a row-major [row][k] buffer (ld in elems,
// multiple of 8). Lane mapping per CDNA5 ISA 7.12.2 (16-bit A 16x32):
// lanes 0-15: K = [k0,k0+8) and [k0+16,k0+24); lanes 16-31: +8.
__device__ __forceinline__ v16bf load_frag(const unsigned short* base, int row0,
                                           int k0, int ld, int lane) {
    const int hf = lane >> 4, r = lane & 15;
    const unsigned short* p = base + (size_t)(row0 + r) * ld + k0 + hf * 8;
    FragU f;
    f.u[0] = *(const uint4*)p;
    f.u[1] = *(const uint4*)(p + 16);
    return f.v;
}

// ---------------------------------------------------------------- prep kernels
__global__ void k_prep_nodes(const float* __restrict__ nf,
                             unsigned short* __restrict__ s_bf,
                             unsigned short* __restrict__ v_bf) {
    int row = blockIdx.x * blockDim.x + threadIdx.x;
    if (row >= NP) return;
    if (row < N_NODES) {
        const float* src = nf + (size_t)row * 512;
        for (int i = 0; i < 128; ++i) {
            s_bf[(size_t)row * 128 + i] = f2bf(src[i]);
#pragma unroll
            for (int c = 0; c < 3; ++c)
                v_bf[(size_t)c * NP * 128 + (size_t)row * 128 + i] = f2bf(src[128 + 3 * i + c]);
        }
    } else {
        for (int i = 0; i < 128; ++i) {
            s_bf[(size_t)row * 128 + i] = 0;
            for (int c = 0; c < 3; ++c)
                v_bf[(size_t)c * NP * 128 + (size_t)row * 128 + i] = 0;
        }
    }
}

// src is (K,N) fp32 row-major -> dst is (N,Kpad) bf16 row-major (W^T, K zero-padded)
__global__ void k_transW(const float* __restrict__ src, unsigned short* __restrict__ dst,
                         int K, int N, int Kpad) {
    int idx = blockIdx.x * blockDim.x + threadIdx.x;
    if (idx >= N * Kpad) return;
    int n = idx / Kpad, k = idx - n * Kpad;
    dst[idx] = (k < K) ? f2bf(src[(size_t)k * N + n]) : (unsigned short)0;
}

__global__ void k_cvt_bf(const float* __restrict__ src, unsigned short* __restrict__ dst, int n) {
    int i = blockIdx.x * blockDim.x + threadIdx.x;
    if (i < n) dst[i] = f2bf(src[i]);
}

// ------------------------------------------------------------- node-side GEMMs
// Computes sc_s/sc_v (-> d_out), up_s/up_v (fp32 ws), down (bf16 ws).
// 68 16x16 output tiles per 16-node block, distributed over 4 waves.
__global__ void __launch_bounds__(128) k_node_gemms(
    const unsigned short* __restrict__ s_bf,
    const unsigned short* __restrict__ v_bf,
    const unsigned short* __restrict__ Wt_skip0,
    const unsigned short* __restrict__ Wt_skip1,
    const unsigned short* __restrict__ Wt_up0,
    const unsigned short* __restrict__ Wt_up1,
    const unsigned short* __restrict__ Wt_down,
    float* __restrict__ up_s,
    float* __restrict__ up_v,
    unsigned short* __restrict__ down_bf,
    float* __restrict__ sc_out) {
    const int lane = threadIdx.x & 31;
    const int wave = threadIdx.x >> 5;
    const int row0 = blockIdx.x * 16;
    const float sC = 0.08838834764831845f;  // 1/sqrt(128)

    for (int t = wave; t < 68; t += 4) {
        const unsigned short *A, *B;
        int nt, c = 0, mode;
        if (t < 8)       { A = s_bf; B = Wt_skip0; nt = t;      mode = 0; }
        else if (t < 16) { A = s_bf; B = Wt_up0;   nt = t - 8;  mode = 1; }
        else if (t < 20) { A = s_bf; B = Wt_down;  nt = t - 16; mode = 2; }
        else if (t < 44) { c = (t - 20) >> 3; nt = (t - 20) & 7;
                           A = v_bf + (size_t)c * NP * 128; B = Wt_skip1; mode = 3; }
        else             { c = (t - 44) >> 3; nt = (t - 44) & 7;
                           A = v_bf + (size_t)c * NP * 128; B = Wt_up1;   mode = 4; }

        v8f acc = {};
#pragma unroll
        for (int k0 = 0; k0 < 128; k0 += 32) {
            v16bf a = load_frag(A, row0, k0, 128, lane);
            v16bf b = load_frag(B, nt * 16, k0, 128, lane);
            acc = __builtin_amdgcn_wmma_f32_16x16x32_bf16(false, a, false, b,
                                                          (short)0, acc, false, false);
        }
        const int col = lane & 15, hf = lane >> 4;
#pragma unroll
        for (int r = 0; r < 8; ++r) {
            int row = row0 + hf * 8 + r;
            if (row >= N_NODES) continue;
            float x = acc[r] * sC;
            int j = nt * 16 + col;
            if      (mode == 0) sc_out[(size_t)row * 512 + j] = x;
            else if (mode == 1) up_s[(size_t)row * 128 + j] = x;
            else if (mode == 2) down_bf[(size_t)row * 64 + j] = f2bf(x);
            else if (mode == 3) sc_out[(size_t)row * 512 + 128 + j * 3 + c] = x;
            else                up_v[(size_t)c * N_NODES * 128 + (size_t)row * 128 + j] = x;
        }
    }
}

// ---------------------------------------------- fused edge MLP + tensor product
// 64 edges per block, 2 waves; each wave owns TWO 16-row M-tiles so every
// B (weight) fragment load feeds two WMMAs (halves L2 weight traffic and gives
// two independent WMMA chains for latency hiding).
#define MT     64
#define AUG_LD 168    // 160 + pad (16B-aligned rows)
#define H_LD   264    // 256 + pad

__device__ __forceinline__ void mlp_layer2(const unsigned short* src, int src_ld, int kpad,
                                           const unsigned short* W, int nout, float scale,
                                           unsigned short* dst, int dst_ld, int lane) {
    const int col = lane & 15, hf = lane >> 4;
    for (int nt = 0; nt < (nout >> 4); ++nt) {
        v8f acc0 = {}, acc1 = {};
        for (int k0 = 0; k0 < kpad; k0 += 32) {
            v16bf b  = load_frag(W, nt * 16, k0, kpad, lane);
            v16bf a0 = load_frag(src, 0,  k0, src_ld, lane);
            v16bf a1 = load_frag(src, 16, k0, src_ld, lane);
            acc0 = __builtin_amdgcn_wmma_f32_16x16x32_bf16(false, a0, false, b,
                                                           (short)0, acc0, false, false);
            acc1 = __builtin_amdgcn_wmma_f32_16x16x32_bf16(false, a1, false, b,
                                                           (short)0, acc1, false, false);
        }
#pragma unroll
        for (int r = 0; r < 8; ++r) {
            float x0 = siluf(acc0[r] * scale);
            float x1 = siluf(acc1[r] * scale);
            dst[(hf * 8 + r) * dst_ld + nt * 16 + col] = f2bf(x0);
            dst[(16 + hf * 8 + r) * dst_ld + nt * 16 + col] = f2bf(x1);
        }
    }
}

__global__ void __launch_bounds__(64) k_edge_mlp_tp(
    const float* __restrict__ edge_feats,
    const float* __restrict__ edge_attrs,
    const int*   __restrict__ edge_index,
    const unsigned short* __restrict__ down_bf,
    const float* __restrict__ up_s,
    const float* __restrict__ up_v,
    const unsigned short* __restrict__ Wt_m0,
    const unsigned short* __restrict__ Wt_m1,
    const unsigned short* __restrict__ Wt_m2,
    const unsigned short* __restrict__ Wt_m3,
    float* __restrict__ m0,
    float* __restrict__ m1) {
    __shared__ __align__(16) unsigned short sh_aug[MT * AUG_LD];
    __shared__ __align__(16) unsigned short sh_h0[MT * H_LD];
    __shared__ __align__(16) unsigned short sh_h1[MT * H_LD];
    __shared__ int   sh_snd[MT], sh_rcv[MT];
    __shared__ float sh_y[MT][4];

    const int lane = threadIdx.x & 31;
    const int wave = threadIdx.x >> 5;
    const int e0 = blockIdx.x * MT;

    {   // stage aug (bf16) + per-edge scalars; each thread builds one full row,
        // and waves only touch their own 32 rows (threads 32w..32w+31 -> rows
        // 32w..32w+31), so no cross-wave LDS dependence.
        int row = threadIdx.x;
        int e = e0 + row;                       // N_EDGES % MT == 0 -> no guard
        int snd = edge_index[2 * e], rcv = edge_index[2 * e + 1];
        sh_snd[row] = snd; sh_rcv[row] = rcv;
#pragma unroll
        for (int i = 0; i < 4; ++i) sh_y[row][i] = edge_attrs[4 * e + i];
        for (int col = 0; col < 160; ++col) {
            unsigned short v;
            if (col < 8)        v = f2bf(edge_feats[(size_t)e * 8 + col]);
            else if (col < 72)  v = down_bf[(size_t)snd * 64 + (col - 8)];
            else if (col < 136) v = down_bf[(size_t)rcv * 64 + (col - 72)];
            else                v = 0;
            sh_aug[row * AUG_LD + col] = v;
        }
    }
    __syncthreads();

    const unsigned short* myAug = sh_aug + wave * 32 * AUG_LD;
    unsigned short* myH0 = sh_h0 + wave * 32 * H_LD;
    unsigned short* myH1 = sh_h1 + wave * 32 * H_LD;

    mlp_layer2(myAug, AUG_LD, 160, Wt_m0, 256, 0.08574929257125442f, myH0, H_LD, lane); // /sqrt(136)
    mlp_layer2(myH0, H_LD, 256, Wt_m1, 256, 0.0625f, myH1, H_LD, lane);
    mlp_layer2(myH1, H_LD, 256, Wt_m2, 256, 0.0625f, myH0, H_LD, lane);

    // layer 4 (tpw) fused with tensor product + atomic segment-sum
    const float INVS3 = 0.5773502691896258f;
    const int col = lane & 15, hf = lane >> 4;
    for (int nt = 0; nt < 32; ++nt) {
        v8f acc[2] = {{}, {}};
        for (int k0 = 0; k0 < 256; k0 += 32) {
            v16bf b  = load_frag(Wt_m3, nt * 16, k0, 256, lane);
            v16bf a0 = load_frag(myH0, 0,  k0, H_LD, lane);
            v16bf a1 = load_frag(myH0, 16, k0, H_LD, lane);
            acc[0] = __builtin_amdgcn_wmma_f32_16x16x32_bf16(false, a0, false, b,
                                                             (short)0, acc[0], false, false);
            acc[1] = __builtin_amdgcn_wmma_f32_16x16x32_bf16(false, a1, false, b,
                                                             (short)0, acc[1], false, false);
        }
        int ncol = nt * 16 + col;
        int q = ncol >> 7, j = ncol & 127;
#pragma unroll
        for (int tile = 0; tile < 2; ++tile) {
#pragma unroll
            for (int r = 0; r < 8; ++r) {
                int rowL = wave * 32 + tile * 16 + hf * 8 + r;
                float tw = acc[tile][r] * 0.0625f;
                int snd = sh_snd[rowL], rcv = sh_rcv[rowL];
                float y0 = sh_y[rowL][0];
                size_t sb = (size_t)snd * 128 + j;
                if (q == 0) {               // w1 * xs * y0 -> m0[:, j]
                    atomicAdd(&m0[(size_t)rcv * 256 + j], tw * up_s[sb] * y0);
                } else if (q == 1) {        // (w2*xs) x yv -> m1[c][:, j]
                    float base = tw * up_s[sb];
#pragma unroll
                    for (int c = 0; c < 3; ++c)
                        atomicAdd(&m1[(size_t)c * NP * 256 + (size_t)rcv * 256 + j],
                                  base * sh_y[rowL][1 + c]);
                } else if (q == 2) {        // w3 * xv * y0 -> m1[c][:, 128+j]
#pragma unroll
                    for (int c = 0; c < 3; ++c) {
                        float xv = up_v[(size_t)c * N_NODES * 128 + sb];
                        atomicAdd(&m1[(size_t)c * NP * 256 + (size_t)rcv * 256 + 128 + j],
                                  tw * xv * y0);
                    }
                } else {                    // w4 * <xv,yv>/sqrt(3) -> m0[:, 128+j]
                    float dot = 0.f;
#pragma unroll
                    for (int c = 0; c < 3; ++c)
                        dot += up_v[(size_t)c * N_NODES * 128 + sb] * sh_y[rowL][1 + c];
                    atomicAdd(&m0[(size_t)rcv * 256 + 128 + j], tw * dot * INVS3);
                }
            }
        }
    }
}

// ------------------------------------------------------------- final linears
__global__ void __launch_bounds__(128) k_final(
    const unsigned short* __restrict__ mbf,       // m0,m1c0,m1c1,m1c2 each NPx256
    const unsigned short* __restrict__ Wt_lin0,
    const unsigned short* __restrict__ Wt_lin1,
    float* __restrict__ out) {
    const int lane = threadIdx.x & 31;
    const int wave = threadIdx.x >> 5;
    const int row0 = blockIdx.x * 16;

    for (int t = wave; t < 32; t += 4) {
        int mat = t >> 3, nt = t & 7;
        const unsigned short* A = mbf + (size_t)mat * NP * 256;
        const unsigned short* B = (mat == 0) ? Wt_lin0 : Wt_lin1;
        v8f acc = {};
#pragma unroll
        for (int k0 = 0; k0 < 256; k0 += 32) {
            v16bf a = load_frag(A, row0, k0, 256, lane);
            v16bf b = load_frag(B, nt * 16, k0, 256, lane);
            acc = __builtin_amdgcn_wmma_f32_16x16x32_bf16(false, a, false, b,
                                                          (short)0, acc, false, false);
        }
        const int col = lane & 15, hf = lane >> 4;
#pragma unroll
        for (int r = 0; r < 8; ++r) {
            int row = row0 + hf * 8 + r;
            if (row >= N_NODES) continue;
            // out[n, j, slot]; slot 0 = msg_s, 1..3 = msg_v components
            out[(size_t)row * 512 + (nt * 16 + col) * 4 + mat] = acc[r] * 0.00390625f; // /(16*16)
        }
    }
}

// ---------------------------------------------------------------- ws layout
static constexpr size_t OFF_SBF  = 0;
static constexpr size_t SZ_SBF   = (size_t)NP * 128 * 2;
static constexpr size_t OFF_VBF  = OFF_SBF + SZ_SBF;
static constexpr size_t SZ_VBF   = 3 * SZ_SBF;
static constexpr size_t OFF_DOWN = OFF_VBF + SZ_VBF;
static constexpr size_t SZ_DOWN  = (size_t)N_NODES * 64 * 2;
static constexpr size_t OFF_UPS  = OFF_DOWN + SZ_DOWN;
static constexpr size_t SZ_UPS   = (size_t)N_NODES * 128 * 4;
static constexpr size_t OFF_UPV  = OFF_UPS + SZ_UPS;
static constexpr size_t SZ_UPV   = 3 * SZ_UPS;
static constexpr size_t OFF_W    = OFF_UPV + SZ_UPV;
static constexpr size_t W_SKIP0 = 0,       W_SKIP1 = 32768,  W_UP0 = 65536,  W_UP1 = 98304;
static constexpr size_t W_DOWN  = 131072,  W_M0 = 147456,    W_M1 = 229376,  W_M2 = 360448;
static constexpr size_t W_M3    = 491520,  W_LIN0 = 753664,  W_LIN1 = 819200, W_END = 884736;
static constexpr size_t OFF_M0  = OFF_W + W_END;
static constexpr size_t SZ_M0   = (size_t)NP * 256 * 4;
static constexpr size_t OFF_M1  = OFF_M0 + SZ_M0;
static constexpr size_t SZ_M1   = 3 * SZ_M0;
static constexpr size_t OFF_MBF = OFF_M1 + SZ_M1;

extern "C" void kernel_launch(void* const* d_in, const int* in_sizes, int n_in,
                              void* d_out, int out_size, void* d_ws, size_t ws_size,
                              hipStream_t stream) {
    (void)in_sizes; (void)n_in; (void)out_size; (void)ws_size;
    const float* node_feats = (const float*)d_in[1];
    const float* edge_attrs = (const float*)d_in[2];
    const float* edge_feats = (const float*)d_in[3];
    const int*   edge_index = (const int*)d_in[4];
    const float* W_up0   = (const float*)d_in[5];
    const float* W_up1   = (const float*)d_in[6];
    const float* W_down  = (const float*)d_in[7];
    const float* Wm0     = (const float*)d_in[8];
    const float* Wm1     = (const float*)d_in[9];
    const float* Wm2     = (const float*)d_in[10];
    const float* Wm3     = (const float*)d_in[11];
    const float* W_lin0  = (const float*)d_in[12];
    const float* W_lin1  = (const float*)d_in[13];
    const float* W_skip0 = (const float*)d_in[14];
    const float* W_skip1 = (const float*)d_in[15];

    char* ws = (char*)d_ws;
    unsigned short* s_bf    = (unsigned short*)(ws + OFF_SBF);
    unsigned short* v_bf    = (unsigned short*)(ws + OFF_VBF);
    unsigned short* down_bf = (unsigned short*)(ws + OFF_DOWN);
    float* up_s = (float*)(ws + OFF_UPS);
    float* up_v = (float*)(ws + OFF_UPV);
    unsigned short* wb = (unsigned short*)(ws + OFF_W);
    unsigned short *Wt_skip0 = wb + W_SKIP0/2, *Wt_skip1 = wb + W_SKIP1/2;
    unsigned short *Wt_up0 = wb + W_UP0/2, *Wt_up1 = wb + W_UP1/2, *Wt_down = wb + W_DOWN/2;
    unsigned short *Wt_m0 = wb + W_M0/2, *Wt_m1 = wb + W_M1/2, *Wt_m2 = wb + W_M2/2;
    unsigned short *Wt_m3 = wb + W_M3/2, *Wt_lin0 = wb + W_LIN0/2, *Wt_lin1 = wb + W_LIN1/2;
    float* m0 = (float*)(ws + OFF_M0);
    float* m1 = (float*)(ws + OFF_M1);
    unsigned short* mbf = (unsigned short*)(ws + OFF_MBF);

    float* out = (float*)d_out;
    float* sc_out = out + (size_t)N_NODES * 512;

    k_prep_nodes<<<(NP + 127) / 128, 128, 0, stream>>>(node_feats, s_bf, v_bf);

    auto T = [&](const float* src, unsigned short* dst, int K, int N, int Kpad) {
        int tot = N * Kpad;
        k_transW<<<(tot + 255) / 256, 256, 0, stream>>>(src, dst, K, N, Kpad);
    };
    T(W_skip0, Wt_skip0, 128, 128, 128);
    T(W_skip1, Wt_skip1, 128, 128, 128);
    T(W_up0,   Wt_up0,   128, 128, 128);
    T(W_up1,   Wt_up1,   128, 128, 128);
    T(W_down,  Wt_down,  128,  64, 128);
    T(Wm0,     Wt_m0,    136, 256, 160);
    T(Wm1,     Wt_m1,    256, 256, 256);
    T(Wm2,     Wt_m2,    256, 256, 256);
    T(Wm3,     Wt_m3,    256, 512, 256);
    T(W_lin0,  Wt_lin0,  256, 128, 256);
    T(W_lin1,  Wt_lin1,  256, 128, 256);

    hipMemsetAsync(ws + OFF_M0, 0, SZ_M0 + SZ_M1, stream);

    k_node_gemms<<<NP / 16, 128, 0, stream>>>(s_bf, v_bf, Wt_skip0, Wt_skip1, Wt_up0,
                                              Wt_up1, Wt_down, up_s, up_v, down_bf, sc_out);

    k_edge_mlp_tp<<<N_EDGES / MT, 64, 0, stream>>>(edge_feats, edge_attrs, edge_index,
                                                   down_bf, up_s, up_v,
                                                   Wt_m0, Wt_m1, Wt_m2, Wt_m3, m0, m1);

    int ntot = 4 * NP * 256;
    k_cvt_bf<<<(ntot + 255) / 256, 256, 0, stream>>>(m0, mbf, ntot);

    k_final<<<NP / 16, 128, 0, stream>>>(mbf, Wt_lin0, Wt_lin1, out);
}
